// NodeSpliter_64656437674056
// MI455X (gfx1250) — matrix operand order repaired and checked by reading, not compiled, not run on hardware
//
#include <hip/hip_runtime.h>
#include <hip/hip_bf16.h>

// ---------------------------------------------------------------------------
// NodeSplitter fused MLP for MI455X (gfx1250, wave32, WMMA f16->f32).
//
// Sizes (fixed by the reference):
//   N=262144 nodes, F=64, G=64, KIN=F+G=128, H=512, B=2, BF=B*F=128,
//   E=512 events, P=256 parents-per-event, PAIRS=P*E=131072, PEB=PAIRS*B=262144
//
// Per block: one M=16 tile of (p,e) pairs, 256 threads = 8 waves.
//   A-tile gathered via global_load_async_to_lds_b128 (ASYNCcnt + barrier).
//   L1: [16x128]@[128x512]+b1, ReLU  (W1 streamed in 4 K-chunks of 32 -> LDS f16)
//   L2: [16x512]@[512x128]+b2       (W2 streamed in 4 K-chunks of 128 -> LDS f16)
//   children written straight to new_x rows N + 2*t + {0,1}.
// ---------------------------------------------------------------------------

typedef __attribute__((ext_vector_type(16))) _Float16 v16h;
typedef __attribute__((ext_vector_type(8)))  float    v8f;
typedef __attribute__((address_space(3))) float lds_f32_t;

#define N_NODES   262144
#define F_DIM     64
#define G_DIM     64
#define KIN       128
#define H_DIM     512
#define BF_DIM    128
#define E_DIM     512
#define P_DIM     256
#define PAIRS     (P_DIM * E_DIM)   // 131072
#define PEB       (PAIRS * 2)       // 262144 (== N_NODES)
#define TILE_M    16

#define A_LD      132               // f32 A-tile row stride (pad: bank offset 4)
#define W1_LD     520               // f16 [32][520] panel (pad vs 512)
#define W2_LD     136               // f16 [128][136] panel (pad vs 128)
#define H_LD      520               // f16 hidden row stride (pad vs 512)

// ---------------------------------------------------------------------------
// Main fused MLP kernel
// ---------------------------------------------------------------------------
__global__ __launch_bounds__(256)
void node_split_mlp(const float* __restrict__ x,
                    const float* __restrict__ gf,
                    const int*   __restrict__ event,
                    const int*   __restrict__ pidx,
                    const float* __restrict__ W1,
                    const float* __restrict__ b1,
                    const float* __restrict__ W2,
                    const float* __restrict__ b2,
                    float* __restrict__ out_x)
{
    __shared__ float    lds_Af[TILE_M][A_LD];     //  8.4 KB gathered inputs (f32)
    __shared__ _Float16 lds_W[128 * W2_LD];       // 34.0 KB streamed weight panel
    __shared__ _Float16 lds_H[TILE_M][H_LD];      // 16.3 KB hidden activations

    const int tid  = threadIdx.x;
    const int lane = tid & 31;
    const int wave = tid >> 5;
    const int t0   = blockIdx.x * TILE_M;         // first (p,e) pair of this tile

    // ---- Gather A tile asynchronously: 16 rows x 128 cols =
    //      [x[parent] | gf[event[parent]]], straight into LDS (no VGPR bounce).
    //      512 float4 chunks, 2 per thread, tracked on ASYNCcnt. ----
    {
        #pragma unroll
        for (int half = 0; half < 2; ++half) {
            const int q = tid + half * 256;       // float4 index 0..511
            const int r = q >> 5;                 // row 0..15 (32 float4 per row)
            const int c = (q & 31) * 4;           // col 0..124 (stays in x or gf)
            const int parent = pidx[t0 + r];
            const float* src;
            if (c < F_DIM) {
                src = x + (size_t)parent * F_DIM + c;
            } else {
                const int ev = event[parent];
                src = gf + (size_t)ev * G_DIM + (c - F_DIM);
            }
            lds_f32_t* dst = (lds_f32_t*)&lds_Af[r][c];
            asm volatile("global_load_async_to_lds_b128 %0, %1, off"
                         :: "v"(dst), "v"(src) : "memory");
        }
        asm volatile("s_wait_asynccnt 0x0" ::: "memory");
    }
    __syncthreads();

    // ======================= Layer 1: A @ W1 + b1, ReLU =======================
    v8f acc[4] = {};
    const int nb = wave * 64;                     // this wave's N-slice of H

    for (int kc = 0; kc < 4; ++kc) {              // K chunks of 32 over KIN=128
        // stage W1 rows [kc*32, kc*32+32) x [0,512) -> lds_W as f16 [32][W1_LD]
        const float4* w1v = (const float4*)(W1 + (size_t)kc * 32 * H_DIM);
        #pragma unroll
        for (int it = 0; it < 16; ++it) {
            const int e4  = tid + it * 256;       // 0..4095 float4's
            const int row = e4 >> 7;              // 128 float4 per 512-float row
            const int col = (e4 & 127) * 4;
            const float4 v = w1v[e4];
            _Float16* dst = &lds_W[row * W1_LD + col];
            dst[0] = (_Float16)v.x; dst[1] = (_Float16)v.y;
            dst[2] = (_Float16)v.z; dst[3] = (_Float16)v.w;
        }
        if (kc < 3)   // prefetch next panel into GL2 (global_prefetch_b8)
            __builtin_prefetch(W1 + (size_t)(kc + 1) * 32 * H_DIM + tid * 16, 0, 1);
        __syncthreads();

        // A fragment (16x32 f16, ISA layout): lane M=L%16, half h=L/16,
        // element i -> VGPR j=i/2, b=i&1, K = (j/4)*16 + (j%4)*2 + b + h*8
        v16h afrag;
        {
            const int M = lane & 15, hh = lane >> 4;
            #pragma unroll
            for (int i = 0; i < 16; ++i) {
                const int j = i >> 1, bbit = i & 1;
                const int k = ((j >> 2) << 4) + ((j & 3) << 1) + bbit + (hh << 3);
                afrag[i] = (_Float16)lds_Af[M][kc * 32 + k];
            }
        }
        #pragma unroll
        for (int nt = 0; nt < 4; ++nt) {
            const int n0 = nb + nt * 16;
            // B fragment (32x16 f16): lane%16 = N column, element i = K row (table order)
            v16h bfrag;
            const _Float16* brow = &lds_W[n0 + (lane & 15)];
            #pragma unroll
            for (int i = 0; i < 16; ++i) bfrag[i] = brow[(((i >> 3) << 4) + (i & 7) + ((lane >> 4) << 3)) * W1_LD];
            acc[nt] = __builtin_amdgcn_wmma_f32_16x16x32_f16(
                false, afrag, false, bfrag, (short)0, acc[nt], false, false);
        }
        __syncthreads();                          // before overwriting lds_W
    }

    // bias + ReLU, spill hidden activations to LDS as f16
    {
        const int col = lane & 15, rg = lane >> 4;
        #pragma unroll
        for (int nt = 0; nt < 4; ++nt) {
            const int n0 = nb + nt * 16;
            const float bias = b1[n0 + col];
            #pragma unroll
            for (int r = 0; r < 8; ++r) {
                float v = acc[nt][r] + bias;
                v = v > 0.0f ? v : 0.0f;
                lds_H[r + rg * 8][n0 + col] = (_Float16)v;
            }
        }
    }
    __syncthreads();

    // ======================= Layer 2: H @ W2 + b2 =======================
    v8f acc2 = {};
    const int n0o = wave * 16;                    // this wave's 16 output columns

    for (int kc = 0; kc < 4; ++kc) {              // K chunks of 128 over H=512
        // stage W2 rows [kc*128, +128) x [0,128) -> lds_W as f16 [128][W2_LD]
        const float4* w2v = (const float4*)(W2 + (size_t)kc * 128 * BF_DIM);
        #pragma unroll
        for (int it = 0; it < 16; ++it) {
            const int e4  = tid + it * 256;
            const int row = e4 >> 5;              // 32 float4 per 128-float row
            const int col = (e4 & 31) * 4;
            const float4 v = w2v[e4];
            _Float16* dst = &lds_W[row * W2_LD + col];
            dst[0] = (_Float16)v.x; dst[1] = (_Float16)v.y;
            dst[2] = (_Float16)v.z; dst[3] = (_Float16)v.w;
        }
        if (kc < 3)
            __builtin_prefetch(W2 + (size_t)(kc + 1) * 128 * BF_DIM + tid * 16, 0, 1);
        __syncthreads();

        #pragma unroll
        for (int ks = 0; ks < 4; ++ks) {          // 4 x K=32 WMMA steps per chunk
            const int kbase = kc * 128 + ks * 32;
            v16h afrag, bfrag;
            const int M = lane & 15, hh = lane >> 4;
            #pragma unroll
            for (int i = 0; i < 16; ++i) {
                const int j = i >> 1, bbit = i & 1;
                const int k = ((j >> 2) << 4) + ((j & 3) << 1) + bbit + (hh << 3);
                afrag[i] = lds_H[M][kbase + k];
            }
            const _Float16* brow = &lds_W[(ks * 32) * W2_LD + n0o + (lane & 15)];
            #pragma unroll
            for (int i = 0; i < 16; ++i) bfrag[i] = brow[(((i >> 3) << 4) + (i & 7) + ((lane >> 4) << 3)) * W2_LD];
            acc2 = __builtin_amdgcn_wmma_f32_16x16x32_f16(
                false, afrag, false, bfrag, (short)0, acc2, false, false);
        }
        __syncthreads();                          // before next W2 panel
    }

    // ---- scatter children: col c -> child b=c/64, feature f=c%64,
    //      new_x row = N + 2*t + b ----
    {
        const int col = lane & 15, rg = lane >> 4;
        const int c = n0o + col;                  // 0..127
        const float bias = b2[c];
        const int child = c >> 6, f = c & 63;
        #pragma unroll
        for (int r = 0; r < 8; ++r) {
            const int M = r + rg * 8;
            const size_t t   = (size_t)(t0 + M);
            const size_t row = (size_t)N_NODES + 2 * t + (size_t)child;
            out_x[row * F_DIM + f] = acc2[r] + bias;
        }
    }
}

// ---------------------------------------------------------------------------
// Copy x -> new_x[0:N] (float4 vectorized)
// ---------------------------------------------------------------------------
__global__ __launch_bounds__(256)
void copy_x_kernel(const float* __restrict__ x, float* __restrict__ out)
{
    const size_t i = (size_t)blockIdx.x * 256 + threadIdx.x;   // float4 index
    ((float4*)out)[i] = ((const float4*)x)[i];
}

// ---------------------------------------------------------------------------
// Bookkeeping: new_event, new_edges, new_edge_attr (written as float).
// N_NODES == PEB, so one grid covers both the event head and all metadata.
// ---------------------------------------------------------------------------
__global__ __launch_bounds__(256)
void meta_kernel(const int* __restrict__ event, const int* __restrict__ pidx,
                 float* __restrict__ ev_out, float* __restrict__ edges_out,
                 float* __restrict__ attr_out)
{
    const int i = blockIdx.x * 256 + threadIdx.x;   // 0..PEB-1
    // head: new_event[0:N] = event
    ev_out[i] = (float)event[i];
    // tail: ev = tile(repeat(arange(E), B), P)
    const int within = i & (E_DIM * 2 - 1);         // i % (E*B)
    const int e = within >> 1;
    const int p = i >> 10;                          // i / (E*B)
    ev_out[(size_t)N_NODES + i] = (float)e;
    // edges: row0 = parent repeated B times, row1 = N + i
    edges_out[i] = (float)pidx[p * E_DIM + e];
    edges_out[(size_t)PEB + i] = (float)(N_NODES + i);
    // attr: ones
    attr_out[i] = 1.0f;
}

// ---------------------------------------------------------------------------
extern "C" void kernel_launch(void* const* d_in, const int* in_sizes, int n_in,
                              void* d_out, int out_size, void* d_ws, size_t ws_size,
                              hipStream_t stream)
{
    (void)in_sizes; (void)n_in; (void)out_size; (void)d_ws; (void)ws_size;

    const float* x     = (const float*)d_in[0];
    const float* gf    = (const float*)d_in[1];
    const int*   event = (const int*)d_in[2];
    const int*   pidx  = (const int*)d_in[3];
    const float* W1    = (const float*)d_in[4];
    const float* b1    = (const float*)d_in[5];
    const float* W2    = (const float*)d_in[6];
    const float* b2    = (const float*)d_in[7];

    float* out       = (float*)d_out;
    float* out_x     = out;                                        // (N+PEB)*F
    float* out_event = out + (size_t)(N_NODES + PEB) * F_DIM;      // N+PEB
    float* out_edges = out_event + (size_t)(N_NODES + PEB);        // 2*PEB
    float* out_attr  = out_edges + (size_t)2 * PEB;                // PEB

    // 1) copy x into new_x head: N*F/4 float4 elements
    hipLaunchKernelGGL(copy_x_kernel,
                       dim3((N_NODES * F_DIM / 4) / 256), dim3(256), 0, stream,
                       x, out_x);

    // 2) bookkeeping outputs
    hipLaunchKernelGGL(meta_kernel,
                       dim3(PEB / 256), dim3(256), 0, stream,
                       event, pidx, out_event, out_edges, out_attr);

    // 3) fused gather + MLP + child scatter
    hipLaunchKernelGGL(node_split_mlp,
                       dim3(PAIRS / TILE_M), dim3(256), 0, stream,
                       x, gf, event, pidx, W1, b1, W2, b2, out_x);
}
